// LPKTPlus_11493332484347
// MI455X (gfx1250) — compile-verified, matrix operand order checked by reading, not run on hardware
//
#include <hip/hip_runtime.h>
#include <hip/hip_bf16.h>

// ---------------------------------------------------------------------------
// LPKT-style model on gfx1250: WMMA f32_16x16x32_f16 for every GEMM.
// Phase 1: f32->f16 weight/embedding prep (+ fold correct-broadcast cols of W1)
// Phase 2: per-position precompute (learning_emb, it_emb, diff, disc)
// Phase 3: sequential 511-step scan; 16 batch rows per WG, h kept in WMMA
//          accumulator registers; W4/Wa fragments register-resident; W2/W3
//          K-loop software-pipelined; next-step tiles prefetched.
// ---------------------------------------------------------------------------

typedef __attribute__((ext_vector_type(16))) _Float16 v16h;
typedef __attribute__((ext_vector_type(8)))  _Float16 v8h;
typedef __attribute__((ext_vector_type(8)))  float    v8f;

#define B_  256
#define S_  512
#define DK_ 128
#define DC_ 50
#define C_  128
#define NQ_ 10000
#define NIT_ 1024
#define NUT_ 1024

__device__ __forceinline__ float sig_(float x) { return 1.0f / (1.0f + __expf(-x)); }
__device__ __forceinline__ float th_(float x)  { float e = __expf(2.0f * x); return (e - 1.0f) / (e + 1.0f); }

__device__ __forceinline__ v8f vz8() {
    v8f z;
#pragma unroll
    for (int i = 0; i < 8; ++i) z[i] = 0.0f;
    return z;
}
__device__ __forceinline__ v8h hz8() {
    v8h z;
#pragma unroll
    for (int i = 0; i < 8; ++i) z[i] = (_Float16)0.0f;
    return z;
}

// Load one 16-lane-striped 16x32 (A) / 32x16 (B) f16 fragment.
// For A: row = m = lane%16.  For B (row-major weight [N][K]): row = n-local.
// Per lane the 16 halves are two contiguous 8-half chunks:
//   k = kbase + 8*(lane>=16) + {0..7}  and  k = kbase + 16 + 8*(lane>=16) + {0..7}
__device__ __forceinline__ v16h frag_ld(const _Float16* base, int row, int stride, int kbase) {
    int lane = threadIdx.x & 31;
    int lg   = lane >> 4;
    const _Float16* p = base + row * stride + kbase + 8 * lg;
    union { v16h v; struct { v8h lo, hi; } s; } u;
    u.s.lo = *(const v8h*)(p);
    u.s.hi = *(const v8h*)(p + 16);
    return u.v;
}

__device__ __forceinline__ v8f wmma_(v16h a, v16h b, v8f c) {
    return __builtin_amdgcn_wmma_f32_16x16x32_f16(false, a, false, b, (short)0, c, false, false);
}

// ---------------------------------------------------------------------------
// Phase 1 kernels
// ---------------------------------------------------------------------------
__global__ void k_cvt_f32_f16(const float* __restrict__ src, _Float16* __restrict__ dst, int n) {
    int i = blockIdx.x * blockDim.x + threadIdx.x;
    if (i < n) dst[i] = (_Float16)src[i];
}

// Split W1 [DK][DC+2*DK] row-major into W1q (cols 0:128), W1u (cols 128:256),
// and fold the 50 broadcast-correct columns into w1c[d] = sum_j W1[d][256+j].
__global__ void k_w1_prep(const float* __restrict__ W1,
                          _Float16* __restrict__ W1q, _Float16* __restrict__ W1u,
                          float* __restrict__ w1c) {
    int idx = blockIdx.x * blockDim.x + threadIdx.x;   // 0 .. 128*128-1
    int d = idx >> 7, k = idx & 127;
    const int RS = DC_ + 2 * DK_;                      // 306
    W1q[idx] = (_Float16)W1[d * RS + k];
    W1u[idx] = (_Float16)W1[d * RS + DK_ + k];
    if (idx < DK_) {
        float s = 0.0f;
        for (int j = 0; j < DC_; ++j) s += W1[idx * RS + 2 * DK_ + j];
        w1c[idx] = s;
    }
}

// ---------------------------------------------------------------------------
// Phase 2: per-position precompute.  16 positions per WG, 8 waves, each wave
// owns one 16-wide N-tile of the 128 output dims.
// ---------------------------------------------------------------------------
__global__ void k_pre(const int* __restrict__ qseq, const int* __restrict__ itseq,
                      const int* __restrict__ utseq, const int* __restrict__ cseq,
                      const _Float16* __restrict__ Eq16, const _Float16* __restrict__ Eut16,
                      const _Float16* __restrict__ Eit16,
                      const _Float16* __restrict__ W1q, const _Float16* __restrict__ W1u,
                      const float* __restrict__ w1c, const float* __restrict__ b1,
                      const _Float16* __restrict__ Wd16, const float* __restrict__ bd,
                      const float* __restrict__ Wdisc, const float* __restrict__ bdisc,
                      _Float16* __restrict__ learnbuf, _Float16* __restrict__ itbuf,
                      _Float16* __restrict__ diffbuf, float* __restrict__ discbuf) {
    __shared__ __align__(16) _Float16 qt[16 * 128];
    __shared__ __align__(16) _Float16 utt[16 * 128];
    __shared__ __align__(16) _Float16 ot[16 * 128];
    __shared__ int   qid[16], utid[16], itid[16];
    __shared__ float corrL[16];

    const int p0  = blockIdx.x * 16;
    const int tid = threadIdx.x;
    const int r = tid >> 4, c = tid & 15;
    const int lane = tid & 31, nl = lane & 15, lg = lane >> 4, nt = tid >> 5;
    const int n = nt * 16 + nl;

    if (tid < 16) {
        int p = p0 + tid;
        qid[tid]   = qseq[p];
        utid[tid]  = utseq[p];
        itid[tid]  = itseq[p];
        corrL[tid] = (float)cseq[p];
    }
    __syncthreads();

    // Gather embedding rows: q & ut into LDS; it straight through to global f16.
    *(v8h*)&qt[r * 128 + c * 8]  = *(const v8h*)&Eq16[(size_t)qid[r] * 128 + c * 8];
    *(v8h*)&utt[r * 128 + c * 8] = *(const v8h*)&Eut16[(size_t)utid[r] * 128 + c * 8];
    {
        v8h iv = *(const v8h*)&Eit16[(size_t)itid[r] * 128 + c * 8];
        *(v8h*)&itbuf[(size_t)(p0 + r) * 128 + c * 8] = iv;
    }
    __syncthreads();

    // learning = q @ W1q^T + ut @ W1u^T + corr*w1c + b1
    v8f acc = vz8();
#pragma unroll
    for (int kt = 0; kt < 4; ++kt)
        acc = wmma_(frag_ld(qt, nl, 128, kt * 32), frag_ld(W1q, n, 128, kt * 32), acc);
#pragma unroll
    for (int kt = 0; kt < 4; ++kt)
        acc = wmma_(frag_ld(utt, nl, 128, kt * 32), frag_ld(W1u, n, 128, kt * 32), acc);
    {
        float bb = b1[n], wc = w1c[n];
#pragma unroll
        for (int i = 0; i < 8; ++i) {
            int m = i + 8 * lg;
            ot[m * 128 + n] = (_Float16)(acc[i] + bb + corrL[m] * wc);
        }
    }
    __syncthreads();
    *(v8h*)&learnbuf[(size_t)(p0 + r) * 128 + c * 8] = *(const v8h*)&ot[r * 128 + c * 8];
    __syncthreads();

    // diff = sigmoid(q @ Wd^T + bd)
    acc = vz8();
#pragma unroll
    for (int kt = 0; kt < 4; ++kt)
        acc = wmma_(frag_ld(qt, nl, 128, kt * 32), frag_ld(Wd16, n, 128, kt * 32), acc);
    {
        float bb = bd[n];
#pragma unroll
        for (int i = 0; i < 8; ++i) {
            int m = i + 8 * lg;
            ot[m * 128 + n] = (_Float16)sig_(acc[i] + bb);
        }
    }
    __syncthreads();
    *(v8h*)&diffbuf[(size_t)(p0 + r) * 128 + c * 8] = *(const v8h*)&ot[r * 128 + c * 8];

    // disc = sigmoid(q . wdisc + bdisc) * 10
    if (tid < 16) {
        float s = 0.0f;
        for (int k = 0; k < DK_; ++k) s += (float)qt[tid * 128 + k] * Wdisc[k];
        discbuf[p0 + tid] = sig_(s + bdisc[0]) * 10.0f;
    }
}

// ---------------------------------------------------------------------------
// Phase 3: sequential scan.  16 batch rows per WG; 511 steps; h lives in the
// WMMA accumulator register distribution (lane holds column n, rows r+8*lg).
// zbuf row layout (f16, stride 512): [learning_pre | it | learning | h]
// z4  row layout (f16, stride 384): [h | LG | it]
// ---------------------------------------------------------------------------
__global__ void k_scan(const _Float16* __restrict__ learnbuf, const _Float16* __restrict__ itbuf,
                       const _Float16* __restrict__ diffbuf, const float* __restrict__ discbuf,
                       const float* __restrict__ h0,
                       const _Float16* __restrict__ W2f, const float* __restrict__ b2,
                       const _Float16* __restrict__ W3f, const float* __restrict__ b3,
                       const _Float16* __restrict__ W4f, const float* __restrict__ b4,
                       const _Float16* __restrict__ Waf, const float* __restrict__ ba,
                       float* __restrict__ out) {
    __shared__ __align__(16) _Float16 zbuf[16 * 512];
    __shared__ __align__(16) _Float16 z4[16 * 384];
    __shared__ __align__(16) _Float16 dbuf[16 * 128];
    __shared__ float dscL[16];
    __shared__ float ysum[16];

    const int m0  = blockIdx.x * 16;
    const int tid = threadIdx.x;
    const int r = tid >> 4, c = tid & 15;
    const int lane = tid & 31, nl = lane & 15, lg = lane >> 4, nt = tid >> 5;
    const int n = nt * 16 + nl;

    // Loop-invariant per-wave weight fragments: W4 (12) + Wa (4) in registers.
    v16h w4r[12];
#pragma unroll
    for (int kt = 0; kt < 12; ++kt) w4r[kt] = frag_ld(W4f, n, 384, kt * 32);
    v16h war[4];
#pragma unroll
    for (int kt = 0; kt < 4; ++kt) war[kt] = frag_ld(Waf, n, 128, kt * 32);

    // Per-wave loop-invariant bias scalars.
    const float bb2 = b2[n], bb3 = b3[n], bb4 = b4[n], bban = ba[n];

    // init: learning_pre = 0, pred[:,0] = 0, h = h0 (f32 regs + f16 mirrors)
    *(v8h*)&zbuf[r * 512 + c * 8] = hz8();
    if (tid < 16) out[(size_t)(m0 + tid) * S_] = 0.0f;

    v8f h;
#pragma unroll
    for (int i = 0; i < 8; ++i) h[i] = h0[(size_t)(m0 + i + 8 * lg) * DK_ + n];
#pragma unroll
    for (int i = 0; i < 8; ++i) {
        int m = i + 8 * lg;
        _Float16 hv = (_Float16)h[i];
        zbuf[m * 512 + 384 + n] = hv;
        z4[m * 384 + n]         = hv;
    }
    __syncthreads();

    for (int t = 0; t < S_ - 1; ++t) {
        // --- stage per-step tiles ---
        size_t rowbase = ((size_t)(m0 + r) * S_ + t) * DK_;
        {
            v8h lv = *(const v8h*)&learnbuf[rowbase + c * 8];
            *(v8h*)&zbuf[r * 512 + 256 + c * 8] = lv;
            v8h iv = *(const v8h*)&itbuf[rowbase + c * 8];
            *(v8h*)&zbuf[r * 512 + 128 + c * 8] = iv;
            *(v8h*)&z4[r * 384 + 256 + c * 8]   = iv;
            v8h dv = *(const v8h*)&diffbuf[rowbase + DK_ + c * 8];   // s = t+1
            *(v8h*)&dbuf[r * 128 + c * 8] = dv;
        }
        if (tid < 16) {
            dscL[tid] = discbuf[(size_t)(m0 + tid) * S_ + t + 1];
            ysum[tid] = 0.0f;
        }
        // Prefetch next step's tiles (L2 -> near cache) while we compute.
        if (t + 1 < S_ - 1) {
            __builtin_prefetch(&learnbuf[rowbase + DK_ + c * 8], 0, 3);
            __builtin_prefetch(&itbuf[rowbase + DK_ + c * 8], 0, 3);
            __builtin_prefetch(&diffbuf[rowbase + 2 * DK_ + c * 8], 0, 3);
        }
        __syncthreads();

        // --- learning_gain / gamma_l:  z(16x512) @ {W2,W3}^T ---
        // Software-pipelined K loop: fetch kt+1 fragments before kt's WMMAs.
        v8f aG = vz8(), aL = vz8();
        {
            v16h a_cur = frag_ld(zbuf, nl, 512, 0);
            v16h b_cur = frag_ld(W2f, n, 512, 0);
            v16h c_cur = frag_ld(W3f, n, 512, 0);
#pragma unroll 4
            for (int kt = 0; kt < 16; ++kt) {
                int ktn = (kt + 1) & 15;
                v16h a_nxt = frag_ld(zbuf, nl, 512, ktn * 32);
                v16h b_nxt = frag_ld(W2f, n, 512, ktn * 32);
                v16h c_nxt = frag_ld(W3f, n, 512, ktn * 32);
                aG = wmma_(a_cur, b_cur, aG);
                aL = wmma_(a_cur, c_cur, aL);
                a_cur = a_nxt; b_cur = b_nxt; c_cur = c_nxt;
            }
        }
        v8f LG;
        {
#pragma unroll
            for (int i = 0; i < 8; ++i) {
                float g  = th_(aG[i] + bb2);
                float gl = sig_(aL[i] + bb3);
                LG[i] = gl * (g + 1.0f) * 0.5f;
                z4[(i + 8 * lg) * 384 + 128 + n] = (_Float16)LG[i];
            }
        }
        __syncthreads();

        // --- gamma_f:  [h_pre | LG | it](16x384) @ W4^T;  h = LG + gf*h ---
        v8f a4 = vz8();
#pragma unroll
        for (int kt = 0; kt < 12; ++kt)
            a4 = wmma_(frag_ld(z4, nl, 384, kt * 32), w4r[kt], a4);
        {
#pragma unroll
            for (int i = 0; i < 8; ++i) {
                float gf = sig_(a4[i] + bb4);
                h[i] = LG[i] + gf * h[i];
                int m = i + 8 * lg;
                _Float16 hv = (_Float16)h[i];
                zbuf[m * 512 + 384 + n] = hv;
                z4[m * 384 + n]         = hv;
            }
        }
        __syncthreads();

        // --- ability = sigmoid(h @ Wa^T + ba);  y = sigmoid(g * sum((a-d)*d)) ---
        v8f aa = vz8();
#pragma unroll
        for (int kt = 0; kt < 4; ++kt)
            aa = wmma_(frag_ld(zbuf + 384, nl, 512, kt * 32), war[kt], aa);
        {
            float part[8];
#pragma unroll
            for (int i = 0; i < 8; ++i) {
                float ab = sig_(aa[i] + bban);
                float d  = (float)dbuf[(i + 8 * lg) * 128 + n];
                part[i] = (ab - d) * d;
            }
            // reduce over the 16 lanes of each half-wave (fixed m per half)
            for (int off = 1; off < 16; off <<= 1) {
#pragma unroll
                for (int i = 0; i < 8; ++i) part[i] += __shfl_xor(part[i], off, 32);
            }
            if (nl == 0) {
#pragma unroll
                for (int i = 0; i < 8; ++i) atomicAdd(&ysum[i + 8 * lg], part[i]);
            }
        }
        __syncthreads();
        if (tid < 16)
            out[(size_t)(m0 + tid) * S_ + t + 1] = sig_(dscL[tid] * ysum[tid]);

        // learning_pre <- learning (next step's z)
        {
            v8h cp = *(const v8h*)&zbuf[r * 512 + 256 + c * 8];
            *(v8h*)&zbuf[r * 512 + c * 8] = cp;
        }
        __syncthreads();
    }
}

// ---------------------------------------------------------------------------
// Host side
// ---------------------------------------------------------------------------
extern "C" void kernel_launch(void* const* d_in, const int* in_sizes, int n_in,
                              void* d_out, int out_size, void* d_ws, size_t ws_size,
                              hipStream_t stream) {
    const int*   qseq  = (const int*)d_in[0];
    const int*   itseq = (const int*)d_in[1];
    const int*   utseq = (const int*)d_in[2];
    const int*   cseq  = (const int*)d_in[3];
    const float* h0    = (const float*)d_in[4];
    const float* E_q   = (const float*)d_in[5];
    const float* E_ut  = (const float*)d_in[6];
    const float* E_it  = (const float*)d_in[7];
    const float* W1    = (const float*)d_in[8];
    const float* b1    = (const float*)d_in[9];
    const float* W2    = (const float*)d_in[10];
    const float* b2    = (const float*)d_in[11];
    const float* W3    = (const float*)d_in[12];
    const float* b3    = (const float*)d_in[13];
    const float* W4    = (const float*)d_in[14];
    const float* b4    = (const float*)d_in[15];
    const float* Wa    = (const float*)d_in[16];
    const float* ba    = (const float*)d_in[17];
    const float* Wd    = (const float*)d_in[18];
    const float* bd    = (const float*)d_in[19];
    const float* Wdisc = (const float*)d_in[20];
    const float* bdisc = (const float*)d_in[21];
    float* out = (float*)d_out;

    // ---- workspace carve (256B aligned regions) ----
    char*  base = (char*)d_ws;
    size_t off  = 0;
    auto carve = [&](size_t bytes) -> void* {
        off = (off + 255) & ~(size_t)255;
        void* p = base + off;
        off += bytes;
        return p;
    };
    _Float16* Eq16  = (_Float16*)carve((size_t)(NQ_ + 1) * DK_ * 2);
    _Float16* Eut16 = (_Float16*)carve((size_t)(NUT_ + 1) * DK_ * 2);
    _Float16* Eit16 = (_Float16*)carve((size_t)(NIT_ + 1) * DK_ * 2);
    _Float16* W1q16 = (_Float16*)carve((size_t)DK_ * DK_ * 2);
    _Float16* W1u16 = (_Float16*)carve((size_t)DK_ * DK_ * 2);
    float*    w1c   = (float*)carve((size_t)DK_ * 4);
    _Float16* W2f   = (_Float16*)carve((size_t)DK_ * 4 * DK_ * 2);
    _Float16* W3f   = (_Float16*)carve((size_t)DK_ * 4 * DK_ * 2);
    _Float16* W4f   = (_Float16*)carve((size_t)DK_ * 3 * DK_ * 2);
    _Float16* Waf   = (_Float16*)carve((size_t)C_ * DK_ * 2);
    _Float16* Wd16  = (_Float16*)carve((size_t)C_ * DK_ * 2);
    _Float16* learnbuf = (_Float16*)carve((size_t)B_ * S_ * DK_ * 2);
    _Float16* itbuf    = (_Float16*)carve((size_t)B_ * S_ * DK_ * 2);
    _Float16* diffbuf  = (_Float16*)carve((size_t)B_ * S_ * C_ * 2);
    float*    discbuf  = (float*)carve((size_t)B_ * S_ * 4);
    (void)ws_size; (void)in_sizes; (void)n_in; (void)out_size;

    const int TB = 256;
    auto blk = [](int n, int tb) { return (n + tb - 1) / tb; };

    // ---- phase 1: f32 -> f16 conversions ----
    k_cvt_f32_f16<<<blk((NQ_ + 1) * DK_, TB), TB, 0, stream>>>(E_q,  Eq16,  (NQ_ + 1) * DK_);
    k_cvt_f32_f16<<<blk((NUT_ + 1) * DK_, TB), TB, 0, stream>>>(E_ut, Eut16, (NUT_ + 1) * DK_);
    k_cvt_f32_f16<<<blk((NIT_ + 1) * DK_, TB), TB, 0, stream>>>(E_it, Eit16, (NIT_ + 1) * DK_);
    k_cvt_f32_f16<<<blk(DK_ * 4 * DK_, TB), TB, 0, stream>>>(W2, W2f, DK_ * 4 * DK_);
    k_cvt_f32_f16<<<blk(DK_ * 4 * DK_, TB), TB, 0, stream>>>(W3, W3f, DK_ * 4 * DK_);
    k_cvt_f32_f16<<<blk(DK_ * 3 * DK_, TB), TB, 0, stream>>>(W4, W4f, DK_ * 3 * DK_);
    k_cvt_f32_f16<<<blk(C_ * DK_, TB), TB, 0, stream>>>(Wa, Waf, C_ * DK_);
    k_cvt_f32_f16<<<blk(C_ * DK_, TB), TB, 0, stream>>>(Wd, Wd16, C_ * DK_);
    k_w1_prep<<<blk(DK_ * DK_, TB), TB, 0, stream>>>(W1, W1q16, W1u16, w1c);

    // ---- phase 2: per-position precompute (8192 WGs x 256 threads) ----
    k_pre<<<(B_ * S_) / 16, TB, 0, stream>>>(qseq, itseq, utseq, cseq,
                                             Eq16, Eut16, Eit16,
                                             W1q16, W1u16, w1c, b1,
                                             Wd16, bd, Wdisc, bdisc,
                                             learnbuf, itbuf, diffbuf, discbuf);

    // ---- phase 3: sequential scan (16 WGs x 256 threads, 511 steps) ----
    k_scan<<<B_ / 16, TB, 0, stream>>>(learnbuf, itbuf, diffbuf, discbuf, h0,
                                       W2f, b2, W3f, b3, W4f, b4, Waf, ba, out);
}
